// _SwinTransformerBlock_893353197803
// MI455X (gfx1250) — matrix operand order repaired and checked by reading, not compiled, not run on hardware
//
#include <hip/hip_runtime.h>
#include <hip/hip_bf16.h>

typedef __attribute__((ext_vector_type(16))) _Float16 v16h;
typedef __attribute__((ext_vector_type(8)))  _Float16 v8h;
typedef __attribute__((ext_vector_type(8)))  float    v8f;
typedef __attribute__((ext_vector_type(4)))  unsigned u32x4;
typedef __attribute__((ext_vector_type(8)))  int      i32x8;
typedef __attribute__((ext_vector_type(4)))  int      i32x4;

// ---- problem constants -----------------------------------------------------
constexpr int  kB   = 64;
constexpr int  kH   = 56;
constexpr int  kW   = 56;
constexpr int  kC   = 96;
constexpr int  kNH  = 3;
constexpr int  kWS  = 7;
constexpr int  kSS  = 3;
constexpr int  kN   = kWS * kWS;               // 49 tokens / window
constexpr int  kNW  = (kH / kWS) * (kW / kWS); // 64 windows / image
constexpr int  kHD  = kC / kNH;                // 32
constexpr int  kHID = 4 * kC;                  // 384
constexpr unsigned kM = (unsigned)kB * kNW * kN; // 200704 tokens
constexpr float kSCALE = 0.17677669529663687f;   // 32^-0.5

#if defined(__has_builtin)
#if __has_builtin(__builtin_amdgcn_tensor_load_to_lds) && __has_builtin(__builtin_amdgcn_s_wait_tensorcnt)
#define USE_TDM 1
#endif
#endif
#ifndef USE_TDM
#define USE_TDM 0
#endif

// ---- WMMA fragment loaders -------------------------------------------------
// A-matrix 16x32 f16: lane<16 holds K {0..15}, lane>=16
// holds K {16..31}; caller pre-offsets pointer by hi*16.
__device__ __forceinline__ v16h ld_a_frag(const _Float16* p) {
  v8h lo = *(const v8h*)(p);
  v8h hi = *(const v8h*)(p + 8);
  return __builtin_shufflevector(lo, hi, 0,1,2,3,4,5,6,7,8,9,10,11,12,13,14,15);
}
// B-matrix 32x16 f16: lane holds one column, contiguous K; caller pre-offsets
// pointer by hi*16 -> 16 contiguous halves.
__device__ __forceinline__ v16h ld_b_frag(const _Float16* p) {
  v8h lo = *(const v8h*)(p);
  v8h hi = *(const v8h*)(p + 8);
  return __builtin_shufflevector(lo, hi, 0,1,2,3,4,5,6,7,8,9,10,11,12,13,14,15);
}

// NB 16x16 tiles (16 x NB*16) of A(MxK,f16) * W(NxK,f16)^T; A-frag reused
// across the NB weight tiles (6x less A traffic on this memory-bound block).
template<int K, int NB>
__device__ __forceinline__ void gemm_tiles(const _Float16* __restrict__ A,
                                           const _Float16* __restrict__ W,
                                           unsigned m0, unsigned n0, int lane,
                                           v8f acc[NB]) {
  const int lr = lane & 15;
  const int hi = lane >> 4;
  const _Float16* ap = A + (size_t)(m0 + lr) * K + hi * 16;
  const _Float16* wp = W + (size_t)(n0 + lr) * K + hi * 16;
#pragma unroll
  for (int nb = 0; nb < NB; ++nb) acc[nb] = (v8f){0.f,0.f,0.f,0.f,0.f,0.f,0.f,0.f};
#pragma unroll
  for (int k0 = 0; k0 < K; k0 += 32) {
    __builtin_prefetch(ap + k0 + 128, 0, 3);
    v16h a = ld_a_frag(ap + k0);
#pragma unroll
    for (int nb = 0; nb < NB; ++nb) {
      v16h b = ld_b_frag(wp + (size_t)nb * 16 * K + k0);
      acc[nb] = __builtin_amdgcn_wmma_f32_16x16x32_f16(false, a, false, b,
                                                       (short)0, acc[nb], false, false);
    }
  }
}

// ---- kernel 0: f32 -> f16 weight cast --------------------------------------
__global__ void cvt_f16_kernel(const float* __restrict__ s,
                               _Float16* __restrict__ d, int n) {
  int i = blockIdx.x * 256 + threadIdx.x;
  if (i < n) d[i] = (_Float16)s[i];
}

// ---- kernels 1/5: LayerNorm (+ optional shift/window partition) ------------
template<bool WINDOWED>
__global__ __launch_bounds__(256)
void ln_kernel(const float* __restrict__ x, const float* __restrict__ g,
               const float* __restrict__ b, _Float16* __restrict__ out) {
  const int lane = threadIdx.x & 31;
  const unsigned tokw = blockIdx.x * 8u + (threadIdx.x >> 5);
  unsigned src;
  if (WINDOWED) {
    unsigned win = tokw / kN, t = tokw - win * kN;
    unsigned bi = win / kNW, w = win - bi * kNW;
    unsigned wy = w >> 3, wx = w & 7;
    unsigned iy = t / kWS, ix = t - iy * kWS;
    unsigned sy = (wy * kWS + iy + kSS) % kH;  // roll(x,-SS): h[i]=x[(i+SS)%H]
    unsigned sx = (wx * kWS + ix + kSS) % kW;
    src = (bi * (kH * kW) + sy * kW + sx) * kC;
  } else {
    src = tokw * kC;
  }
  float v0 = x[src + lane], v1 = x[src + lane + 32], v2 = x[src + lane + 64];
  float s = v0 + v1 + v2;
  float q = v0 * v0 + v1 * v1 + v2 * v2;
#pragma unroll
  for (int m = 1; m < 32; m <<= 1) {
    s += __shfl_xor(s, m, 32);
    q += __shfl_xor(q, m, 32);
  }
  float mean = s * (1.f / 96.f);
  float var  = q * (1.f / 96.f) - mean * mean;
  float rstd = rsqrtf(var + 1e-5f);
  unsigned o = tokw * kC;
  out[o + lane]      = (_Float16)((v0 - mean) * rstd * g[lane]      + b[lane]);
  out[o + lane + 32] = (_Float16)((v1 - mean) * rstd * g[lane + 32] + b[lane + 32]);
  out[o + lane + 64] = (_Float16)((v2 - mean) * rstd * g[lane + 64] + b[lane + 64]);
}

// ---- kernel 2: QKV GEMM (M x 288); wave = 16 rows x 96 cols (one of q/k/v) -
__global__ __launch_bounds__(256)
void qkv_kernel(const _Float16* __restrict__ A, const _Float16* __restrict__ W,
                const float* __restrict__ bias, _Float16* __restrict__ qb,
                _Float16* __restrict__ kb, _Float16* __restrict__ vb) {
  const int lane = threadIdx.x & 31;
  const unsigned tile = blockIdx.x * 8u + (threadIdx.x >> 5);
  const unsigned mt = tile / 3u, g = tile - mt * 3u;   // g: 0=q 1=k 2=v
  const unsigned m0 = mt * 16u;
  v8f acc[6];
  gemm_tiles<kC, 6>(A, W, m0, g * 96u, lane, acc);
  const int hi8 = (lane >> 4) * 8;
  const int lr = lane & 15;
  _Float16* dst = g == 0 ? qb : (g == 1 ? kb : vb);
  const float scl = (g == 0) ? kSCALE : 1.f;
  float bbv[6];
#pragma unroll
  for (int nt = 0; nt < 6; ++nt) bbv[nt] = bias[g * 96u + nt * 16 + lr];
#pragma unroll
  for (int j = 0; j < 8; ++j) {
    unsigned m = m0 + j + hi8;
    unsigned win = m / kN, tok = m - win * kN;
    unsigned base = (win * kNH * kN + tok) * kHD;   // +head*49*32 below
#pragma unroll
    for (int nt = 0; nt < 6; ++nt) {
      int nl = nt * 16 + lr;           // 0..95
      int head = nl >> 5, hd = nl & 31;
      dst[base + (unsigned)head * (kN * kHD) + hd] = (_Float16)((acc[nt][j] + bbv[nt]) * scl);
    }
  }
}

// ---- kernel 3: attention per (window, head) --------------------------------
__global__ __launch_bounds__(128)
void attn_kernel(const _Float16* __restrict__ qg, const _Float16* __restrict__ kg,
                 const _Float16* __restrict__ vg, const float* __restrict__ rpb,
                 _Float16* __restrict__ aout) {
  __shared__ __align__(16) _Float16 qs[64 * 32];
  __shared__ __align__(16) _Float16 ks[64 * 32];
  __shared__ __align__(16) _Float16 vs[64 * 32];
  __shared__ __align__(16) _Float16 sP[4 * 16 * 64];

  const int wh = blockIdx.x;
  const int win = wh / kNH, head = wh - win * kNH;
  const int w = win & (kNW - 1);
  const int wy7 = (w >> 3) * kWS, wx7 = (w & 7) * kWS;
  const int t = threadIdx.x;
  const unsigned elbase = (unsigned)(win * kNH + head) * (kN * kHD); // halves

#if USE_TDM
  // Tensor Data Mover: 2D tile 32x64 over a 32x49 tensor; rows 49..63 read
  // OOB -> zero, giving the zero padding for free. One descriptor per matrix.
  if (t < 32) {
    const _Float16* gsrc[3] = {qg + elbase, kg + elbase, vg + elbase};
    _Float16* lds[3] = {qs, ks, vs};
#pragma unroll
    for (int mtx = 0; mtx < 3; ++mtx) {
      unsigned long long ga = (unsigned long long)(uintptr_t)gsrc[mtx];
      unsigned la = (unsigned)(uintptr_t)lds[mtx];
      u32x4 g0;
      g0[0] = 1u;                                   // count=1, user descriptor
      g0[1] = la;                                   // lds_addr (bytes)
      g0[2] = (unsigned)ga;                         // global_addr[31:0]
      g0[3] = (unsigned)((ga >> 32) & 0x1FFFFFFu) | 0x80000000u; // [56:32]|type=2
      i32x8 g1;
      g1[0] = 0x00010000;    // workgroup_mask=0, data_size=1 (2B)
      g1[1] = 32 << 16;      // tensor_dim0 = 32 (HD)
      g1[2] = 49 << 16;      // tensor_dim1 = 49 (tokens)
      g1[3] = 32 << 16;      // tile_dim0 = 32
      g1[4] = 64;            // tile_dim1 = 64 (pad rows OOB -> zero)
      g1[5] = 32;            // tensor_dim0_stride = 32
      g1[6] = 0;
      g1[7] = 0;
      i32x4 gz4 = {0, 0, 0, 0};
      i32x8 gz8 = {0, 0, 0, 0, 0, 0, 0, 0};
      __builtin_amdgcn_tensor_load_to_lds(g0, g1, gz4, gz4, gz8, 0);
    }
    __builtin_amdgcn_s_wait_tensorcnt(0);
  }
#else
  {
    const unsigned* qsrc = (const unsigned*)(qg + elbase);
    const unsigned* ksrc = (const unsigned*)(kg + elbase);
    const unsigned* vsrc = (const unsigned*)(vg + elbase);
    unsigned* qd = (unsigned*)qs; unsigned* kd = (unsigned*)ks; unsigned* vd = (unsigned*)vs;
#pragma unroll
    for (int i = 0; i < 8; ++i) {
      int d = i * 128 + t;
      bool ok = d < kN * kHD / 2;   // 784 dwords
      qd[d] = ok ? qsrc[d] : 0u;
      kd[d] = ok ? ksrc[d] : 0u;
      vd[d] = ok ? vsrc[d] : 0u;
    }
  }
#endif
  __syncthreads();

  const int lane = t & 31, wid = t >> 5;
  const int lr = lane & 15, hi = lane >> 4;
  const int m0 = wid * 16;

  // S = (q*SCALE) @ k^T : 16x64 per wave, K = HD = 32
  v16h aq = ld_a_frag(qs + (m0 + lr) * 32 + hi * 16);
  v8f S[4];
#pragma unroll
  for (int nt = 0; nt < 4; ++nt) {
    v16h bk = ld_b_frag(ks + (nt * 16 + lr) * 32 + hi * 16);
    v8f z = {0.f, 0.f, 0.f, 0.f, 0.f, 0.f, 0.f, 0.f};
    S[nt] = __builtin_amdgcn_wmma_f32_16x16x32_f16(false, aq, false, bk,
                                                   (short)0, z, false, false);
  }

  // rel-pos bias + shift-window mask + row softmax
#pragma unroll
  for (int j = 0; j < 8; ++j) {
    const int r  = m0 + j + hi * 8;
    const int rr = r < kN ? r : kN - 1;
    const int ry = rr / kWS, rx = rr - ry * kWS;
    const int yR = wy7 + ry, xR = wx7 + rx;
    const int lR = (yR < 49 ? 0 : (yR < 53 ? 1 : 2)) * 3 +
                   (xR < 49 ? 0 : (xR < 53 ? 1 : 2));
    float vv[4];
#pragma unroll
    for (int nt = 0; nt < 4; ++nt) {
      int c = nt * 16 + lr;
      if (c < kN) {
        int cy = c / kWS, cx = c - cy * kWS;
        int idx = (ry - cy + 6) * 13 + (rx - cx + 6);
        float bias = rpb[idx * kNH + head];
        int yC = wy7 + cy, xC = wx7 + cx;
        int lC = (yC < 49 ? 0 : (yC < 53 ? 1 : 2)) * 3 +
                 (xC < 49 ? 0 : (xC < 53 ? 1 : 2));
        vv[nt] = S[nt][j] + bias + ((lR != lC) ? -100.f : 0.f);
      } else {
        vv[nt] = -3.0e38f;
      }
    }
    float mx = fmaxf(fmaxf(vv[0], vv[1]), fmaxf(vv[2], vv[3]));
#pragma unroll
    for (int s = 1; s < 16; s <<= 1) mx = fmaxf(mx, __shfl_xor(mx, s, 32));
    float p[4], sm = 0.f;
#pragma unroll
    for (int nt = 0; nt < 4; ++nt) { p[nt] = __expf(vv[nt] - mx); sm += p[nt]; }
#pragma unroll
    for (int s = 1; s < 16; s <<= 1) sm += __shfl_xor(sm, s, 32);
    float inv = 1.f / sm;
#pragma unroll
    for (int nt = 0; nt < 4; ++nt)
      sP[wid * 1024 + (j + hi * 8) * 64 + nt * 16 + lr] = (_Float16)(p[nt] * inv);
  }
  __syncthreads();

  // O = P @ v : 16x32 per wave, K = 64 padded tokens
#pragma unroll
  for (int ot = 0; ot < 2; ++ot) {
    v8f co = {0.f, 0.f, 0.f, 0.f, 0.f, 0.f, 0.f, 0.f};
#pragma unroll
    for (int k0 = 0; k0 < 64; k0 += 32) {
      v16h pa = ld_a_frag(sP + wid * 1024 + lr * 64 + k0 + hi * 16);
      v16h bv;
#pragma unroll
      for (int i = 0; i < 16; ++i)
        bv[i] = vs[(k0 + hi * 16 + i) * 32 + ot * 16 + lr];
      co = __builtin_amdgcn_wmma_f32_16x16x32_f16(false, pa, false, bv,
                                                  (short)0, co, false, false);
    }
#pragma unroll
    for (int j = 0; j < 8; ++j) {
      int tok = m0 + j + hi * 8;
      if (tok < kN)
        aout[((unsigned)win * kN + tok) * kC + head * kHD + ot * 16 + lr] = (_Float16)co[j];
    }
  }
}

// ---- kernel 4: proj GEMM + window reverse + un-shift + residual ------------
__global__ __launch_bounds__(256)
void proj_kernel(const _Float16* __restrict__ A, const _Float16* __restrict__ W,
                 const float* __restrict__ bias, const float* __restrict__ xin,
                 float* __restrict__ x2) {
  const int lane = threadIdx.x & 31;
  const unsigned mt = blockIdx.x * 8u + (threadIdx.x >> 5);
  const unsigned m0 = mt * 16u;
  v8f acc[6];
  gemm_tiles<kC, 6>(A, W, m0, 0u, lane, acc);
  const int hi8 = (lane >> 4) * 8;
  const int lr = lane & 15;
  float bbv[6];
#pragma unroll
  for (int nt = 0; nt < 6; ++nt) bbv[nt] = bias[nt * 16 + lr];
#pragma unroll
  for (int j = 0; j < 8; ++j) {
    unsigned m = m0 + j + hi8;
    unsigned win = m / kN, tok = m - win * kN;
    unsigned bi = win / kNW, w = win - bi * kNW;
    unsigned wy = w >> 3, wx = w & 7;
    unsigned iy = tok / kWS, ix = tok - iy * kWS;
    unsigned y  = (wy * kWS + iy + kSS) % kH;   // reverse roll by +SS
    unsigned xx = (wx * kWS + ix + kSS) % kW;
    unsigned d = (bi * (kH * kW) + y * kW + xx) * kC + lr;
#pragma unroll
    for (int nt = 0; nt < 6; ++nt)
      x2[d + nt * 16] = xin[d + nt * 16] + acc[nt][j] + bbv[nt];
  }
}

// ---- kernel 6: FC1 GEMM + exact GELU ---------------------------------------
__global__ __launch_bounds__(256)
void fc1_kernel(const _Float16* __restrict__ A, const _Float16* __restrict__ W,
                const float* __restrict__ bias, _Float16* __restrict__ o1) {
  const int lane = threadIdx.x & 31;
  const unsigned tile = blockIdx.x * 8u + (threadIdx.x >> 5);
  const unsigned mt = tile >> 2, g = tile & 3u;
  const unsigned m0 = mt * 16u, n0 = g * 96u;
  v8f acc[6];
  gemm_tiles<kC, 6>(A, W, m0, n0, lane, acc);
  const int hi8 = (lane >> 4) * 8;
  const int lr = lane & 15;
  float bbv[6];
#pragma unroll
  for (int nt = 0; nt < 6; ++nt) bbv[nt] = bias[n0 + nt * 16 + lr];
#pragma unroll
  for (int j = 0; j < 8; ++j) {
    unsigned row = (m0 + j + hi8) * kHID + n0 + lr;
#pragma unroll
    for (int nt = 0; nt < 6; ++nt) {
      float v = acc[nt][j] + bbv[nt];
      v = 0.5f * v * (1.f + erff(v * 0.70710678118654752f));
      o1[row + nt * 16] = (_Float16)v;
    }
  }
}

// ---- kernel 7: FC2 GEMM + residual -> d_out --------------------------------
__global__ __launch_bounds__(256)
void fc2_kernel(const _Float16* __restrict__ A, const _Float16* __restrict__ W,
                const float* __restrict__ bias, const float* __restrict__ x2,
                float* __restrict__ out) {
  const int lane = threadIdx.x & 31;
  const unsigned mt = blockIdx.x * 8u + (threadIdx.x >> 5);
  const unsigned m0 = mt * 16u;
  v8f acc[6];
  gemm_tiles<kHID, 6>(A, W, m0, 0u, lane, acc);
  const int hi8 = (lane >> 4) * 8;
  const int lr = lane & 15;
  float bbv[6];
#pragma unroll
  for (int nt = 0; nt < 6; ++nt) bbv[nt] = bias[nt * 16 + lr];
#pragma unroll
  for (int j = 0; j < 8; ++j) {
    unsigned d = (m0 + j + hi8) * kC + lr;
#pragma unroll
    for (int nt = 0; nt < 6; ++nt)
      out[d + nt * 16] = x2[d + nt * 16] + acc[nt][j] + bbv[nt];
  }
}

// ---- workspace layout (bytes, all 256-aligned) -----------------------------
constexpr size_t OFF_WQKV = 0;                        // 288*96*2   = 55296
constexpr size_t OFF_WPRJ = 55296;                    // 96*96*2    = 18432
constexpr size_t OFF_WFC1 = 73728;                    // 384*96*2   = 73728
constexpr size_t OFF_WFC2 = 147456;                   // 96*384*2   = 73728
constexpr size_t SZ_TOKC16 = (size_t)kM * kC * 2;     // 38,535,168
constexpr size_t OFF_HWIN = 221184;                   // aliased: hwin->aout->h2
constexpr size_t OFF_Q    = OFF_HWIN + SZ_TOKC16;
constexpr size_t OFF_K    = OFF_Q + SZ_TOKC16;
constexpr size_t OFF_V    = OFF_K + SZ_TOKC16;
constexpr size_t OFF_X2   = OFF_V + SZ_TOKC16;        // f32, 77,070,336
constexpr size_t OFF_A1   = OFF_X2 + (size_t)kM * kC * 4; // f16 M x 384

extern "C" void kernel_launch(void* const* d_in, const int* in_sizes, int n_in,
                              void* d_out, int out_size, void* d_ws, size_t ws_size,
                              hipStream_t stream) {
  (void)in_sizes; (void)n_in; (void)out_size; (void)ws_size;
  const float* x      = (const float*)d_in[0];
  const float* n1g    = (const float*)d_in[1];
  const float* n1b    = (const float*)d_in[2];
  const float* qkv_w  = (const float*)d_in[3];
  const float* qkv_b  = (const float*)d_in[4];
  const float* rpb    = (const float*)d_in[5];
  const float* proj_w = (const float*)d_in[6];
  const float* proj_b = (const float*)d_in[7];
  const float* n2g    = (const float*)d_in[8];
  const float* n2b    = (const float*)d_in[9];
  const float* fc1_w  = (const float*)d_in[10];
  const float* fc1_b  = (const float*)d_in[11];
  const float* fc2_w  = (const float*)d_in[12];
  const float* fc2_b  = (const float*)d_in[13];
  float* out = (float*)d_out;
  char* ws = (char*)d_ws;

  _Float16* w_qkv = (_Float16*)(ws + OFF_WQKV);
  _Float16* w_prj = (_Float16*)(ws + OFF_WPRJ);
  _Float16* w_fc1 = (_Float16*)(ws + OFF_WFC1);
  _Float16* w_fc2 = (_Float16*)(ws + OFF_WFC2);
  _Float16* hwin  = (_Float16*)(ws + OFF_HWIN);   // also aout / h2 (sequential reuse)
  _Float16* qb    = (_Float16*)(ws + OFF_Q);
  _Float16* kb    = (_Float16*)(ws + OFF_K);
  _Float16* vb    = (_Float16*)(ws + OFF_V);
  float*    x2    = (float*)(ws + OFF_X2);
  _Float16* a1    = (_Float16*)(ws + OFF_A1);

  // 0: weight casts f32 -> f16
  cvt_f16_kernel<<<(288 * 96 + 255) / 256, 256, 0, stream>>>(qkv_w, w_qkv, 288 * 96);
  cvt_f16_kernel<<<(96 * 96 + 255) / 256, 256, 0, stream>>>(proj_w, w_prj, 96 * 96);
  cvt_f16_kernel<<<(384 * 96 + 255) / 256, 256, 0, stream>>>(fc1_w, w_fc1, 384 * 96);
  cvt_f16_kernel<<<(96 * 384 + 255) / 256, 256, 0, stream>>>(fc2_w, w_fc2, 96 * 384);

  const int mtiles = (int)(kM / 16);   // 12544
  // 1: LN1 + cyclic shift + window partition -> hwin (M x 96, f16)
  ln_kernel<true><<<(int)(kM / 8), 256, 0, stream>>>(x, n1g, n1b, hwin);
  // 2: QKV GEMM; wave = 16 rows x 96 cols (entire q, k, or v slice)
  qkv_kernel<<<mtiles * 3 / 8, 256, 0, stream>>>(hwin, w_qkv, qkv_b, qb, kb, vb);
  // 3: attention per (window, head) -> aout (alias hwin)
  attn_kernel<<<kB * kNW * kNH, 128, 0, stream>>>(qb, kb, vb, rpb, hwin);
  // 4: proj GEMM + window reverse + un-shift + residual -> x2 (f32)
  proj_kernel<<<mtiles / 8, 256, 0, stream>>>(hwin, w_prj, proj_b, x, x2);
  // 5: LN2 -> h2 (alias hwin)
  ln_kernel<false><<<(int)(kM / 8), 256, 0, stream>>>(x2, n2g, n2b, hwin);
  // 6: FC1 + GELU -> a1 (M x 384, f16)
  fc1_kernel<<<mtiles * 4 / 8, 256, 0, stream>>>(hwin, w_fc1, fc1_b, a1);
  // 7: FC2 + residual -> out (f32)
  fc2_kernel<<<mtiles / 8, 256, 0, stream>>>(a1, w_fc2, fc2_b, x2, out);
}
